// RAILTDecoder_27513560498694
// MI455X (gfx1250) — matrix operand order repaired and checked by reading, not compiled, not run on hardware
//
#include <hip/hip_runtime.h>
#include <hip/hip_bf16.h>

// ---------------------------------------------------------------------------
// RAILT decoder for MI455X (gfx1250, wave32, WMMA).
// GEMMs: v_wmma_f32_16x16x32_bf16, 32x128 block tile, K-step 64,
// single-barrier double-buffered pipeline, async-to-LDS weight staging
// (ASYNCcnt) when the toolchain exposes the builtin.
// ---------------------------------------------------------------------------

#define D_MODEL 512
#define N_HEADS 8
#define HEAD_D  64
#define CH      50
#define MEMN    16
#define TT      66       // MEM + CH
#define BATCH   16
#define SEQL    2000
#define NCHUNK  40       // SEQL / CH
#define DFF     2048

typedef __attribute__((ext_vector_type(16))) __bf16 v16bf;
typedef __attribute__((ext_vector_type(8)))  __bf16 v8bf;
typedef __attribute__((ext_vector_type(8)))  float  v8f;

enum { ACT_NONE = 0, ACT_RELU = 1, ACT_SIGMOID = 2 };

// ---- CDNA5 async global->LDS copy (ASYNCcnt) with safe fallback -----------
#if __has_builtin(__builtin_amdgcn_global_load_async_to_lds_b128)
  #define GEMM_ASYNC_W 1
  // param type (from clang diagnostic): pointer to 'int __vector_size__(16)'
  typedef int v4i_vs __attribute__((vector_size(16)));
  typedef __attribute__((address_space(1))) v4i_vs* g_v4p;   // global src
  typedef __attribute__((address_space(3))) v4i_vs* l_v4p;   // LDS dst
  #define ASYNC_COPY16(dst, src)                                              \
      __builtin_amdgcn_global_load_async_to_lds_b128(                         \
          (g_v4p)(src), (l_v4p)(dst), 0, 0)
  #if __has_builtin(__builtin_amdgcn_s_wait_asynccnt)
    #define WAIT_ASYNC() __builtin_amdgcn_s_wait_asynccnt(0)
  #else
    #define WAIT_ASYNC() asm volatile("s_wait_asynccnt 0x0" ::: "memory")
  #endif
#else
  #define GEMM_ASYNC_W 0
#endif

__device__ __forceinline__ v8bf pack8(float4 a, float4 b)
{
    v8bf r;
    r[0] = (__bf16)a.x; r[1] = (__bf16)a.y; r[2] = (__bf16)a.z; r[3] = (__bf16)a.w;
    r[4] = (__bf16)b.x; r[5] = (__bf16)b.y; r[6] = (__bf16)b.z; r[7] = (__bf16)b.w;
    return r;
}

// ------------------------------ GEMM (WMMA) --------------------------------
// C[M,N] = act(A[M,K] @ W[N,K]^T + bias[N]) (+ residual[M,N])
// A: f32 row-major (converted to bf16 while staging to LDS)
// W: bf16 row-major [N][K]
// Requires: M % 32 == 0, N % 128 == 0, K % 64 == 0 (true for all call sites).
#define BM 32
#define BN 128
#define KB 64
#define LDSS (KB + 8)   // 72 halfs; 144B rows keep 16B alignment, skew banks

__global__ __launch_bounds__(256)
void gemm_bf16_kernel(const float* __restrict__ A, const __bf16* __restrict__ W,
                      const float* __restrict__ bias, const float* __restrict__ residual,
                      float* __restrict__ C, int M, int N, int K, int act)
{
    __shared__ __align__(16) __bf16 SA[2][BM][LDSS];
    __shared__ __align__(16) __bf16 SW[2][BN][LDSS];

    const int tid   = threadIdx.x;
    const int wave  = tid >> 5;
    const int lane  = tid & 31;
    const int ln16  = lane & 15;
    const bool hi   = lane >= 16;
    const int tileM = blockIdx.y * BM;
    const int tileN = blockIdx.x * BN;
    const int cn    = wave * 16;          // this wave's 16-column strip of BN

    v8f acc[2];
    acc[0] = (v8f){0.f,0.f,0.f,0.f,0.f,0.f,0.f,0.f};
    acc[1] = (v8f){0.f,0.f,0.f,0.f,0.f,0.f,0.f,0.f};

    // staging assignments
    const int arow  = tid >> 3;           // 0..31
    const int acol8 = (tid & 7) * 8;      // 0..56 step 8 (8 f32 each)
    const int wrow  = tid >> 1;           // 0..127
    const int wcol  = (tid & 1) * 32;     // 0 or 32 (32 halfs each)

    const int nk = K / KB;
    const float*  Ag  = A + (size_t)(tileM + arow) * K + acol8;
    const __bf16* Wg  = W + (size_t)(tileN + wrow) * K + wcol;
    const float*  Agp = Ag + KB;          // next-tile prefetch pointers
    const __bf16* Wgp = Wg + KB;

    // ---- prologue: tile 0 in flight ----
    float4 ra0 = *(const float4*)(Ag);
    float4 ra1 = *(const float4*)(Ag + 4);
#if GEMM_ASYNC_W
    #pragma unroll
    for (int j = 0; j < 4; ++j)
        ASYNC_COPY16(&SW[0][wrow][wcol + 8 * j], Wg + 8 * j);
#else
    uint4 rw0 = *(const uint4*)(Wg);
    uint4 rw1 = *(const uint4*)(Wg + 8);
    uint4 rw2 = *(const uint4*)(Wg + 16);
    uint4 rw3 = *(const uint4*)(Wg + 24);
#endif

    int cur = 0;
    for (int k = 0; k < nk; ++k) {
        const int nxt = cur ^ 1;

        // commit A tile k (cvt f32->bf16) and prefetch A tile k+1
        *(v8bf*)&SA[cur][arow][acol8] = pack8(ra0, ra1);
#if !GEMM_ASYNC_W
        *(uint4*)&SW[cur][wrow][wcol +  0] = rw0;
        *(uint4*)&SW[cur][wrow][wcol +  8] = rw1;
        *(uint4*)&SW[cur][wrow][wcol + 16] = rw2;
        *(uint4*)&SW[cur][wrow][wcol + 24] = rw3;
#endif
        if (k + 1 < nk) {
            ra0 = *(const float4*)(Agp);
            ra1 = *(const float4*)(Agp + 4);
            Agp += KB;
#if !GEMM_ASYNC_W
            rw0 = *(const uint4*)(Wgp);
            rw1 = *(const uint4*)(Wgp + 8);
            rw2 = *(const uint4*)(Wgp + 16);
            rw3 = *(const uint4*)(Wgp + 24);
            Wgp += KB;
#endif
        }
#if GEMM_ASYNC_W
        WAIT_ASYNC();          // this wave's W tile-k portion has landed
#endif
        __syncthreads();        // everyone's SA/SW tile-k data is visible
#if GEMM_ASYNC_W
        if (k + 1 < nk) {       // issue W tile k+1 into the other buffer
            #pragma unroll
            for (int j = 0; j < 4; ++j)
                ASYNC_COPY16(&SW[nxt][wrow][wcol + 8 * j], Wgp + 8 * j);
            Wgp += KB;
        }
#endif
        // ---- compute: 2 k32-substeps x 2 M-subtiles = 4 WMMAs ----
        #pragma unroll
        for (int k32 = 0; k32 < 2; ++k32) {
            const int kb32 = k32 * 32;
            // B fragment: lanes 0-15 -> K kb32..kb32+15, lanes 16-31 -> +16
            const __bf16* brp = &SW[cur][cn + ln16][kb32 + (hi ? 16 : 0)];
            v8bf b0 = *(const v8bf*)(brp);
            v8bf b1 = *(const v8bf*)(brp + 8);
            v16bf bfrag = __builtin_shufflevector(b0, b1,
                0,1,2,3,4,5,6,7,8,9,10,11,12,13,14,15);
            #pragma unroll
            for (int m2 = 0; m2 < 2; ++m2) {
                // A fragment: element i -> K = kb32 + (hi?16:0) + i
                const __bf16* arp = &SA[cur][m2 * 16 + ln16][kb32 + (hi ? 16 : 0)];
                v8bf a0 = *(const v8bf*)(arp);
                v8bf a1 = *(const v8bf*)(arp + 8);
                v16bf afrag = __builtin_shufflevector(a0, a1,
                    0,1,2,3,4,5,6,7,8,9,10,11,12,13,14,15);
                acc[m2] = __builtin_amdgcn_wmma_f32_16x16x32_bf16(
                    false, afrag, false, bfrag, (short)0, acc[m2], false, false);
            }
        }
        cur = nxt;
    }

    // D layout: lanes 0-15: M = v, N = lane; lanes 16-31: M = 8+v, N = lane-16
    const int n_ = tileN + cn + ln16;
    const float bv = bias ? bias[n_] : 0.f;
    #pragma unroll
    for (int m2 = 0; m2 < 2; ++m2) {
        const int mbase = tileM + m2 * 16 + (hi ? 8 : 0);
        #pragma unroll
        for (int v = 0; v < 8; ++v) {
            const int m = mbase + v;
            float x = acc[m2][v] + bv;
            if (act == ACT_RELU)         x = x > 0.f ? x : 0.f;
            else if (act == ACT_SIGMOID) x = 1.f / (1.f + __expf(-x));
            if (residual) x += residual[(size_t)m * N + n_];
            C[(size_t)m * N + n_] = x;
        }
    }
}

// ------------------------------ LayerNorm ----------------------------------
// One wave per 512-wide row; 8 rows per 256-thread block.
__global__ __launch_bounds__(256)
void layernorm_kernel(const float* __restrict__ X, const float* __restrict__ g,
                      const float* __restrict__ b, float* __restrict__ Y, int rows)
{
    const int wave = threadIdx.x >> 5;
    const int lane = threadIdx.x & 31;
    const int row  = blockIdx.x * 8 + wave;
    if (row >= rows) return;
    const float* x = X + (size_t)row * D_MODEL;
    float vals[16];
    float s = 0.f, ss = 0.f;
#pragma unroll
    for (int i = 0; i < 16; ++i) {
        const float v = x[lane + i * 32];
        vals[i] = v; s += v; ss += v * v;
    }
#pragma unroll
    for (int off = 16; off > 0; off >>= 1) {
        s  += __shfl_xor(s,  off, 32);
        ss += __shfl_xor(ss, off, 32);
    }
    const float mean = s * (1.f / D_MODEL);
    const float var  = ss * (1.f / D_MODEL) - mean * mean;
    const float inv  = rsqrtf(var + 1e-5f);
    float* y = Y + (size_t)row * D_MODEL;
#pragma unroll
    for (int i = 0; i < 16; ++i) {
        const int c = lane + i * 32;
        y[c] = (vals[i] - mean) * inv * g[c] + b[c];
    }
}

// --------------------------- masked attention ------------------------------
// One block per (b, h). K,V head slices staged in LDS; one wave per query.
// mode 0 (self): masked iff q >= MEM && k > q
// mode 1 (cross): masked iff k >= MEM && (k - MEM) > q
__global__ __launch_bounds__(256)
void attention_kernel(const float* __restrict__ Qb, int qStride,
                      const float* __restrict__ KVb, int kvStride, int kOff, int vOff,
                      float* __restrict__ Ob, int oStride,
                      int Tq, int Tk, int mode)
{
    const int b = blockIdx.x >> 3;
    const int h = blockIdx.x & 7;
    __shared__ float Ks[TT][HEAD_D + 1];
    __shared__ float Vs[TT][HEAD_D + 1];

    for (int idx = threadIdx.x; idx < Tk * HEAD_D; idx += 256) {
        const int t = idx >> 6, d = idx & 63;
        const size_t r = (size_t)(b * Tk + t) * kvStride + h * HEAD_D + d;
        Ks[t][d] = KVb[r + kOff];
        Vs[t][d] = KVb[r + vOff];
    }
    __syncthreads();

    const int wave = threadIdx.x >> 5;
    const int lane = threadIdx.x & 31;
    const float scale = 0.125f;   // 1/sqrt(64)

    for (int q = wave; q < Tq; q += 8) {
        const float* qr = Qb + (size_t)(b * Tq + q) * qStride + h * HEAD_D;
        float sreg[3];
#pragma unroll
        for (int j = 0; j < 3; ++j) {
            const int k = lane + 32 * j;
            float s = -3.0e38f;           // pad: exp underflows to 0
            if (k < Tk) {
                const bool masked = (mode == 0) ? (q >= MEMN && k > q)
                                                : (k >= MEMN && (k - MEMN) > q);
                float a = 0.f;
                for (int d = 0; d < HEAD_D; ++d) a += qr[d] * Ks[k][d];
                s = masked ? -1.0e9f : a * scale;
            }
            sreg[j] = s;
        }
        float mx = fmaxf(fmaxf(sreg[0], sreg[1]), sreg[2]);
#pragma unroll
        for (int off = 16; off > 0; off >>= 1) mx = fmaxf(mx, __shfl_xor(mx, off, 32));
        float sum = 0.f;
#pragma unroll
        for (int j = 0; j < 3; ++j) { sreg[j] = __expf(sreg[j] - mx); sum += sreg[j]; }
#pragma unroll
        for (int off = 16; off > 0; off >>= 1) sum += __shfl_xor(sum, off, 32);
        const float rinv = 1.f / sum;

        float o0 = 0.f, o1 = 0.f;
        const int d0 = 2 * lane, d1 = 2 * lane + 1;
#pragma unroll
        for (int j = 0; j < 3; ++j) {
            for (int k2 = 0; k2 < 32; ++k2) {
                const int k = j * 32 + k2;
                if (k >= Tk) break;                 // uniform across wave
                const float w = __shfl(sreg[j], k2, 32);
                o0 += w * Vs[k][d0];
                o1 += w * Vs[k][d1];
            }
        }
        float* orow = Ob + (size_t)(b * Tq + q) * oStride + h * HEAD_D;
        orow[d0] = o0 * rinv;
        orow[d1] = o1 * rinv;
    }
}

// --------------------------- elementwise helpers ---------------------------
__global__ void f32_to_bf16_kernel(const float* __restrict__ src,
                                   __bf16* __restrict__ dst, int n)
{
    const int i = blockIdx.x * blockDim.x + threadIdx.x;
    if (i < n) dst[i] = (__bf16)src[i];
}

// z[b,t,:] = (t<MEM ? mem[b,t,:] : emb[tok]) + pos[t,:]   (embedding fused)
__global__ void build_z_kernel(const float* __restrict__ mem, const float* __restrict__ emb,
                               const int* __restrict__ in_res, const float* __restrict__ pos,
                               float* __restrict__ z, int ch)
{
    const int row = blockIdx.x;           // b*TT + t, grid = BATCH*TT
    const int b = row / TT, t = row % TT;
    const float* src;
    if (t < MEMN) {
        src = mem + ((size_t)b * MEMN + t) * D_MODEL;
    } else {
        const int tok = in_res[b * SEQL + ch * CH + (t - MEMN)];
        src = emb + (size_t)tok * D_MODEL;
    }
    const float* pp = pos + (size_t)t * D_MODEL;
    float* zp = z + (size_t)row * D_MODEL;
    for (int c = threadIdx.x; c < D_MODEL; c += blockDim.x) zp[c] = src[c] + pp[c];
}

// kv[b,t,:] = t<MEM ? proj_mem[b,t,:] : enc_out[b, ch*CH + t-MEM, :]
__global__ void build_kv_kernel(const float* __restrict__ proj_mem,
                                const float* __restrict__ enc_out,
                                float* __restrict__ kv, int ch)
{
    const int row = blockIdx.x;
    const int b = row / TT, t = row % TT;
    const float* src = (t < MEMN)
        ? proj_mem + ((size_t)b * MEMN + t) * D_MODEL
        : enc_out + ((size_t)b * SEQL + ch * CH + (t - MEMN)) * D_MODEL;
    float* kp = kv + (size_t)row * D_MODEL;
    for (int c = threadIdx.x; c < D_MODEL; c += blockDim.x) kp[c] = src[c];
}

// mem_cand = z[:, :MEM];  h = z[:, MEM:]
__global__ void split_kernel(const float* __restrict__ z,
                             float* __restrict__ mem_cand, float* __restrict__ h)
{
    const int row = blockIdx.x;
    const int b = row / TT, t = row % TT;
    const float* zp = z + (size_t)row * D_MODEL;
    float* dst = (t < MEMN) ? mem_cand + ((size_t)b * MEMN + t) * D_MODEL
                            : h + ((size_t)b * CH + (t - MEMN)) * D_MODEL;
    for (int c = threadIdx.x; c < D_MODEL; c += blockDim.x) dst[c] = zp[c];
}

// mem = first ? mem_cand : g*mem_cand + (1-g)*mem
__global__ void blend_kernel(const float* __restrict__ mem_cand, const float* __restrict__ g,
                             float* __restrict__ mem, int first)
{
    const size_t i = (size_t)blockIdx.x * blockDim.x + threadIdx.x;  // BATCH*MEMN*D elems
    const float c = mem_cand[i];
    mem[i] = first ? c : g[i] * c + (1.f - g[i]) * mem[i];
}

// out[b, ch*CH + t, :] = h[b, t, :]
__global__ void writeout_kernel(const float* __restrict__ h, float* __restrict__ out, int ch)
{
    const int row = blockIdx.x;           // b*CH + t, grid = BATCH*CH
    const int b = row / CH, t = row % CH;
    const float* hp = h + (size_t)row * D_MODEL;
    float* op = out + ((size_t)b * SEQL + ch * CH + t) * D_MODEL;
    for (int c = threadIdx.x; c < D_MODEL; c += blockDim.x) op[c] = hp[c];
}

// ------------------------------- launcher ----------------------------------
static inline size_t align256(size_t x) { return (x + 255) & ~(size_t)255; }

extern "C" void kernel_launch(void* const* d_in, const int* in_sizes, int n_in,
                              void* d_out, int out_size, void* d_ws, size_t ws_size,
                              hipStream_t stream)
{
    // ---- inputs (setup_inputs dict order; nested dicts in insertion order) ----
    const int*   in_res     = (const int*)  d_in[0];
    const float* enc_out    = (const float*)d_in[1];
    const float* enc_mem    = (const float*)d_in[2];
    const float* emb        = (const float*)d_in[3];
    const float* mem_proj_w = (const float*)d_in[4];
    const float* mem_proj_b = (const float*)d_in[5];
    const float* pos        = (const float*)d_in[6];
    const float* sa_in_w    = (const float*)d_in[7];
    const float* sa_in_b    = (const float*)d_in[8];
    const float* sa_out_w   = (const float*)d_in[9];
    const float* sa_out_b   = (const float*)d_in[10];
    const float* ln_attn_g  = (const float*)d_in[11];
    const float* ln_attn_b  = (const float*)d_in[12];
    const float* ffn_w1     = (const float*)d_in[13];
    const float* ffn_b1     = (const float*)d_in[14];
    const float* ffn_w2     = (const float*)d_in[15];
    const float* ffn_b2     = (const float*)d_in[16];
    const float* ln_ffn_g   = (const float*)d_in[17];
    const float* ln_ffn_b   = (const float*)d_in[18];
    const float* emp_w      = (const float*)d_in[19];  // enc_mem_proj_w
    const float* emp_b      = (const float*)d_in[20];
    const float* ca_in_w    = (const float*)d_in[21];
    const float* ca_in_b    = (const float*)d_in[22];
    const float* ca_out_w   = (const float*)d_in[23];
    const float* ca_out_b   = (const float*)d_in[24];
    const float* ln_cq_g    = (const float*)d_in[25];
    const float* ln_cq_b    = (const float*)d_in[26];
    const float* ln_ckv_g   = (const float*)d_in[27];
    const float* ln_ckv_b   = (const float*)d_in[28];
    const float* fc_w1      = (const float*)d_in[29];
    const float* fc_b1      = (const float*)d_in[30];
    const float* fc_w2      = (const float*)d_in[31];
    const float* fc_b2      = (const float*)d_in[32];
    const float* ln_fc_g    = (const float*)d_in[33];
    const float* ln_fc_b    = (const float*)d_in[34];
    const float* gate_w     = (const float*)d_in[35];
    const float* gate_b     = (const float*)d_in[36];
    float* out = (float*)d_out;

    // ---- workspace carve-out (~51 MB) ----
    char* p = (char*)d_ws;
    auto allocF = [&](size_t n) { float*  r = (float*) p; p += align256(n * 4); return r; };
    auto allocH = [&](size_t n) { __bf16* r = (__bf16*)p; p += align256(n * 2); return r; };

    __bf16* w_sa_in   = allocH(1536 * 512);
    __bf16* w_sa_out  = allocH(512 * 512);
    __bf16* w_ffn1    = allocH(2048 * 512);
    __bf16* w_ffn2    = allocH(512 * 2048);
    __bf16* w_ca_in   = allocH(1536 * 512);   // rows 0..511 = Wq, 512..1535 = Wk|Wv
    __bf16* w_ca_out  = allocH(512 * 512);
    __bf16* w_fc1     = allocH(2048 * 512);
    __bf16* w_fc2     = allocH(512 * 2048);
    __bf16* w_gate    = allocH(512 * 512);
    __bf16* w_memp    = allocH(512 * 512);
    __bf16* w_emp     = allocH(512 * 512);

    const int RZ = BATCH * TT;   // 1056
    const int RH = BATCH * CH;   // 800
    const int RM = BATCH * MEMN; // 256

    float* mem      = allocF((size_t)RM * D_MODEL);
    float* proj_mem = allocF((size_t)RM * D_MODEL);
    float* mem_cand = allocF((size_t)RM * D_MODEL);
    float* gbuf     = allocF((size_t)RM * D_MODEL);
    float* z        = allocF((size_t)RZ * D_MODEL);
    float* zn       = allocF((size_t)RZ * D_MODEL);
    float* qkv_s    = allocF((size_t)RZ * 1536);
    float* attn     = allocF((size_t)RZ * D_MODEL);
    float* hid      = allocF((size_t)RZ * DFF);
    float* h        = allocF((size_t)RH * D_MODEL);
    float* tmp800   = allocF((size_t)RH * D_MODEL);   // q LN / fc LN output
    float* kvcat    = allocF((size_t)RZ * D_MODEL);
    float* kvn      = allocF((size_t)RZ * D_MODEL);
    float* q_c      = allocF((size_t)RH * D_MODEL);
    float* kv_c     = allocF((size_t)RZ * 1024);

    auto cvt = [&](const float* s, __bf16* d, int n) {
        f32_to_bf16_kernel<<<(n + 255) / 256, 256, 0, stream>>>(s, d, n);
    };
    auto gemm = [&](const float* A, const __bf16* W, const float* bias, const float* res,
                    float* C, int M, int N, int K, int act) {
        dim3 g(N / BN, M / BM);
        gemm_bf16_kernel<<<g, 256, 0, stream>>>(A, W, bias, res, C, M, N, K, act);
    };
    auto lnorm = [&](const float* X, const float* g_, const float* b_, float* Y, int rows) {
        layernorm_kernel<<<(rows + 7) / 8, 256, 0, stream>>>(X, g_, b_, Y, rows);
    };

    // ---- one-time weight conversion (L2-resident afterwards) ----
    cvt(sa_in_w,  w_sa_in,  1536 * 512);
    cvt(sa_out_w, w_sa_out, 512 * 512);
    cvt(ffn_w1,   w_ffn1,   2048 * 512);
    cvt(ffn_w2,   w_ffn2,   512 * 2048);
    cvt(ca_in_w,  w_ca_in,  1536 * 512);
    cvt(ca_out_w, w_ca_out, 512 * 512);
    cvt(fc_w1,    w_fc1,    2048 * 512);
    cvt(fc_w2,    w_fc2,    512 * 2048);
    cvt(gate_w,   w_gate,   512 * 512);
    cvt(mem_proj_w, w_memp, 512 * 512);
    cvt(emp_w,      w_emp,  512 * 512);

    // ---- pre-loop projections ----
    gemm(enc_mem, w_memp, mem_proj_b, nullptr, mem,      RM, 512, 512, ACT_NONE);
    gemm(enc_mem, w_emp,  emp_b,      nullptr, proj_mem, RM, 512, 512, ACT_NONE);

    // ---- sequential scan over chunks ----
    for (int ch = 0; ch < NCHUNK; ++ch) {
        // z = concat(mem, emb[chunk]) + pos
        build_z_kernel<<<RZ, 256, 0, stream>>>(mem, emb, in_res, pos, z, ch);
        // self-attn block
        lnorm(z, ln_attn_g, ln_attn_b, zn, RZ);
        gemm(zn, w_sa_in, sa_in_b, nullptr, qkv_s, RZ, 1536, 512, ACT_NONE);
        attention_kernel<<<BATCH * N_HEADS, 256, 0, stream>>>(
            qkv_s, 1536, qkv_s, 1536, 512, 1024, attn, 512, TT, TT, 0);
        gemm(attn, w_sa_out, sa_out_b, z, z, RZ, 512, 512, ACT_NONE);
        // FFN block
        lnorm(z, ln_ffn_g, ln_ffn_b, zn, RZ);
        gemm(zn,  w_ffn1, ffn_b1, nullptr, hid, RZ, DFF, 512, ACT_RELU);
        gemm(hid, w_ffn2, ffn_b2, z, z, RZ, 512, DFF, ACT_NONE);
        // split into memory candidate + chunk states
        split_kernel<<<RZ, 256, 0, stream>>>(z, mem_cand, h);
        // cross-attn block
        lnorm(h, ln_cq_g, ln_cq_b, tmp800, RH);
        build_kv_kernel<<<RZ, 256, 0, stream>>>(proj_mem, enc_out, kvcat, ch);
        lnorm(kvcat, ln_ckv_g, ln_ckv_b, kvn, RZ);
        gemm(tmp800, w_ca_in,               ca_in_b,       nullptr, q_c,  RH, 512,  512, ACT_NONE);
        gemm(kvn,    w_ca_in + 512 * 512,   ca_in_b + 512, nullptr, kv_c, RZ, 1024, 512, ACT_NONE);
        attention_kernel<<<BATCH * N_HEADS, 256, 0, stream>>>(
            q_c, 512, kv_c, 1024, 0, 512, attn, 512, CH, TT, 1);
        gemm(attn, w_ca_out, ca_out_b, h, h, RH, 512, 512, ACT_NONE);
        // cross FFN block
        lnorm(h, ln_fc_g, ln_fc_b, tmp800, RH);
        gemm(tmp800, w_fc1, fc_b1, nullptr, hid, RH, DFF, 512, ACT_RELU);
        gemm(hid,    w_fc2, fc_b2, h, h, RH, 512, DFF, ACT_NONE);
        // gated memory update
        gemm(mem_cand, w_gate, gate_b, nullptr, gbuf, RM, 512, 512, ACT_SIGMOID);
        blend_kernel<<<(RM * D_MODEL) / 256, 256, 0, stream>>>(mem_cand, gbuf, mem, ch == 0);
        // emit chunk output
        writeout_kernel<<<RH, 256, 0, stream>>>(h, out, ch);
    }
}
